// Encoder_61899068670101
// MI455X (gfx1250) — matrix operand order repaired and checked by reading, not compiled, not run on hardware
//
#include <hip/hip_runtime.h>

// ---------------------------------------------------------------------------
// Bidirectional LSTM encoder for MI455X (gfx1250, wave32, WMMA).
//   tokens [512,1024] i32, emb [1000,64] f32,
//   Wx_f/Wh_f [64,256] f32, b_f [256] f32, (same _b), out [512,128] f32.
//
// Block = 4 waves, one 16-row batch tile per block. Wave w owns gate-column
// group j=w (tiles {w, w+4, w+8, w+12}) -> 16 v_wmma_f32_16x16x32_f16 per
// step per wave, with ALL 16 weight B-fragments pinned in 128 VGPRs for the
// whole scan (zero weight traffic in steady state). The f16 embedding table
// (125 KB) lives in LDS -- its region is reused from the one-shot weight
// prepack staging -- so the per-step x-gather is 4x ds_load_b128, not global.
// Token loads are software-pipelined two steps deep so their L2 latency is
// hidden behind two full step bodies. h[16,64] is shared across waves via a
// double-buffered LDS tile with one __syncthreads() per timestep.
// Grid = (32 batch tiles) x (2 directions) = 64 blocks.
// ---------------------------------------------------------------------------

typedef __attribute__((ext_vector_type(16))) _Float16 v16h;
typedef __attribute__((ext_vector_type(8)))  _Float16 v8h;
typedef __attribute__((ext_vector_type(8)))  float    v8f;

#define T_STEPS 1024
#define EMB_D   64
#define VOCAB_N 1000
#define NT      16      // 256 gate columns / 16-wide N tiles
#define WAVES   4       // waves per block; wave w = gate-column group j=w

__device__ __forceinline__ float fsig(float x) {
    return 1.0f / (1.0f + __expf(-x));
}
__device__ __forceinline__ float ftanh(float x) {
#if __has_builtin(__builtin_amdgcn_tanhf)
    return __builtin_amdgcn_tanhf(x);
#else
    float e = __expf(2.0f * x);
    return 1.0f - 2.0f / (e + 1.0f);
#endif
}

// Dynamic LDS layout (132096 bytes / block):
//   half  sH[2][16][64]        :   4 KB  double-buffered h staging
//   union region (offset 4096, 32B aligned):
//     prepack phase : v16h sWx[1024], sWh[1024]  (64 KB weight B-fragments)
//     steady  phase : half sEmb[1000][64]        (125 KB f16 embedding table)
extern __shared__ v16h dyn_smem[];

__global__ __launch_bounds__(WAVES * 32) void bilstm_wmma(
    const int*   __restrict__ tokens,   // [512,1024]
    const float* __restrict__ emb,      // [1000,64]
    const float* __restrict__ Wx_f, const float* __restrict__ Wh_f,
    const float* __restrict__ b_f,
    const float* __restrict__ Wx_b, const float* __restrict__ Wh_b,
    const float* __restrict__ b_b,
    float*       __restrict__ out)      // [512,128]
{
    const int dir  = blockIdx.y;                 // 0 = fwd, 1 = bwd
    const float* Wx = dir ? Wx_b : Wx_f;
    const float* Wh = dir ? Wh_b : Wh_f;
    const float* bi = dir ? b_b  : b_f;

    _Float16* sH   = (_Float16*)dyn_smem;                       // [2][16][64]
    v16h*     sWx  = (v16h*)((char*)dyn_smem + 4096);           // [NT*2*32]
    v16h*     sWh  = sWx + NT * 2 * 32;
    _Float16* sEmb = (_Float16*)sWx;                            // [1000][64]

    const int tid     = threadIdx.x;
    const int lane    = tid & 31;
    const int w       = tid >> 5;                // gate-column group j = w
    const int ln      = lane & 15;               // row (A) / col (B,D) in tile
    const int halfsel = lane >> 4;               // half-wave select
    const int b0      = blockIdx.x * 16;         // batch row base (one tile/block)

    // ---- Phase 1: pre-pack Wx/Wh f32 -> f16 B-fragments in LDS ----------
    // B fragment (32x16 f16): lane holds column n = 16*tile + (lane&15);
    // halves e=0..7 hold k = 32*khalf + 8*(lane>>4) + e; halves 8..15 -> +8.
    for (int idx = tid; idx < NT * 2 * 32 * 16; idx += WAVES * 32) {
        int e    = idx & 15;
        int fl   = (idx >> 4) & 31;
        int kh   = (idx >> 9) & 1;
        int tile = idx >> 10;
        int k    = 32 * kh + 8 * (fl >> 4) + e + (e >= 8 ? 8 : 0);
        int col  = 16 * tile + (fl & 15);
        sWx[(tile * 2 + kh) * 32 + fl][e] = (_Float16)Wx[k * 256 + col];
        sWh[(tile * 2 + kh) * 32 + fl][e] = (_Float16)Wh[k * 256 + col];
    }
    // zero the first h buffer (read at step 0)
    for (int idx = tid; idx < 16 * 64; idx += WAVES * 32)
        sH[idx] = (_Float16)0.0f;
    __syncthreads();

    // ---- hoist this wave's 16 weight B-fragments into registers ----------
    // (loop-invariant: 16 x v16h = 128 VGPRs pinned for the whole scan)
    v16h wxB[4][2], whB[4][2];
#pragma unroll
    for (int g = 0; g < 4; ++g) {
        const int tile = 4 * g + w;
        wxB[g][0] = sWx[(tile * 2 + 0) * 32 + lane];
        wxB[g][1] = sWx[(tile * 2 + 1) * 32 + lane];
        whB[g][0] = sWh[(tile * 2 + 0) * 32 + lane];
        whB[g][1] = sWh[(tile * 2 + 1) * 32 + lane];
    }
    __syncthreads();   // everyone hoisted; staging LDS may now be reused

    // ---- Phase 2: f16 embedding table into the reused LDS region --------
    for (int idx = tid; idx < VOCAB_N * EMB_D; idx += WAVES * 32)
        sEmb[idx] = (_Float16)emb[idx];
    __syncthreads();

    // ---- per-lane bias fragment for this wave's 4 gate tiles -------------
    float bfrag[4];
#pragma unroll
    for (int g = 0; g < 4; ++g) bfrag[g] = bi[64 * g + 16 * w + ln];

    // ---- state: lane holds rows m = r + 8*halfsel, col = 16*w + ln -------
    float hreg[8], creg[8];
#pragma unroll
    for (int r = 0; r < 8; ++r) { hreg[r] = 0.0f; creg[r] = 0.0f; }

    const size_t trow = (size_t)(b0 + ln) * T_STEPS;

    // ---- token pipeline, two steps deep ---------------------------------
    // t(s) = dir ? T-1-s : s ; loads for t(s+2) issue at step s and are only
    // waited on two full step bodies later.
    int tokA = tokens[trow + (dir ? (T_STEPS - 1) : 0)];
    int tokB = tokens[trow + (dir ? (T_STEPS - 2) : 1)];

    // =================== serial scan over time ===========================
#pragma unroll 1
    for (int s = 0; s < T_STEPS; ++s) {
        const _Float16* hbufR = sH + (s & 1) * (16 * 64);        // read buf
        _Float16*       hbufW = sH + ((s + 1) & 1) * (16 * 64);  // write buf

        const int tok = tokA;                    // token for this step: ready
        tokA = tokB;
        int sp = s + 2; sp = (sp < T_STEPS) ? sp : (T_STEPS - 1);
        tokB = tokens[trow + (dir ? (T_STEPS - 1 - sp) : sp)];   // in flight

        // mask: row m valid iff tokens[b0+m][t] != 0  (wave ballot, no LDS)
        const unsigned long long bal = __ballot(tok != 0);

        // ---- x_t / h A-fragments from LDS --------------------------------
        // A fragment (16x32 f16): lane holds row m = ln; halves 0..7 ->
        // k = 32*kh + 8*halfsel + e ; halves 8..15 -> +16.
        const _Float16* xr = sEmb + tok * EMB_D;
        const _Float16* hr = hbufR + ln * EMB_D;
        v16h ax[2], ah[2];
#pragma unroll
        for (int kh = 0; kh < 2; ++kh) {
            v8h xlo = *(const v8h*)(xr + 32 * kh + 8 * halfsel);
            v8h xhi = *(const v8h*)(xr + 32 * kh + 16 + 8 * halfsel);
            ax[kh] = __builtin_shufflevector(xlo, xhi,
                      0, 1, 2, 3, 4, 5, 6, 7, 8, 9, 10, 11, 12, 13, 14, 15);
            v8h hlo = *(const v8h*)(hr + 32 * kh + 8 * halfsel);
            v8h hhi = *(const v8h*)(hr + 32 * kh + 16 + 8 * halfsel);
            ah[kh] = __builtin_shufflevector(hlo, hhi,
                      0, 1, 2, 3, 4, 5, 6, 7, 8, 9, 10, 11, 12, 13, 14, 15);
        }

        // ---- this wave's 4 gate tiles: i/f/g/o for col group j = w -------
        // All B operands pinned in registers: 16 back-to-back WMMAs.
        v8f acc[4];
#pragma unroll
        for (int g = 0; g < 4; ++g) {
            v8f a;
            const float bv = bfrag[g];
#pragma unroll
            for (int r = 0; r < 8; ++r) a[r] = bv;
            a = __builtin_amdgcn_wmma_f32_16x16x32_f16(
                    false, ax[0], false, wxB[g][0], (short)0, a, false, false);
            a = __builtin_amdgcn_wmma_f32_16x16x32_f16(
                    false, ax[1], false, wxB[g][1], (short)0, a, false, false);
            a = __builtin_amdgcn_wmma_f32_16x16x32_f16(
                    false, ah[0], false, whB[g][0], (short)0, a, false, false);
            a = __builtin_amdgcn_wmma_f32_16x16x32_f16(
                    false, ah[1], false, whB[g][1], (short)0, a, false, false);
            acc[g] = a;
        }

        // ---- elementwise LSTM cell update (rows m = r + 8*halfsel) -------
#pragma unroll
        for (int r = 0; r < 8; ++r) {
            const float iv = fsig(acc[0][r]);
            const float fv = fsig(acc[1][r]);
            const float gv = ftanh(acc[2][r]);
            const float ov = fsig(acc[3][r]);
            const float cn = fv * creg[r] + iv * gv;
            const float hn = ov * ftanh(cn);
            const int   m  = r + 8 * halfsel;
            if ((bal >> m) & 1ull) { creg[r] = cn; hreg[r] = hn; }
        }

        // ---- stage new h (f16) into the other buffer ---------------------
#pragma unroll
        for (int r = 0; r < 8; ++r)
            hbufW[(r + 8 * halfsel) * 64 + 16 * w + ln] = (_Float16)hreg[r];

        __syncthreads();   // writes visible before next step's reads
    }

    // ---- write final hidden state: out[b][dir*64 + 16*w + ln] ------------
#pragma unroll
    for (int r = 0; r < 8; ++r) {
        const int m = r + 8 * halfsel;
        out[(size_t)(b0 + m) * 128 + dir * 64 + 16 * w + ln] = hreg[r];
    }
}

extern "C" void kernel_launch(void* const* d_in, const int* in_sizes, int n_in,
                              void* d_out, int out_size, void* d_ws, size_t ws_size,
                              hipStream_t stream) {
    (void)in_sizes; (void)n_in; (void)out_size; (void)d_ws; (void)ws_size;
    const int*   tokens = (const int*)  d_in[0];
    const float* emb    = (const float*)d_in[1];
    const float* Wx_f   = (const float*)d_in[2];
    const float* Wh_f   = (const float*)d_in[3];
    const float* b_f    = (const float*)d_in[4];
    const float* Wx_b   = (const float*)d_in[5];
    const float* Wh_b   = (const float*)d_in[6];
    const float* b_b    = (const float*)d_in[7];
    float*       out    = (float*)d_out;

    // 512 batch rows / 16 rows per block = 32 blocks per direction.
    dim3 grid(32, 2), block(WAVES * 32);
    // 4 KB h double-buffer + max(64 KB weight staging, 125 KB f16 emb table)
    size_t lds_bytes = 4096 + (size_t)VOCAB_N * EMB_D * sizeof(_Float16);
    bilstm_wmma<<<grid, block, lds_bytes, stream>>>(
        tokens, emb, Wx_f, Wh_f, b_f, Wx_b, Wh_b, b_b, out);
}
